// SelfAttention_53489522704490
// MI455X (gfx1250) — matrix operand order repaired and checked by reading, not compiled, not run on hardware
//
#include <hip/hip_runtime.h>

// ---------------------------------------------------------------------------
// Stick-breaking (sigmoid-gated) causal attention for MI455X (gfx1250).
//   B=2, S=2048, H=16, D=64, fp32 in/out.
// Flash-style streaming over 32-wide s-tiles in DESCENDING order.
//   att[t,s] = exp( logsigmoid(x) - tail_sum(softplus(x)) ),  rem = exp(-R)
// All exponents <= 0 -> no rescaling. Three WMMA uses per tile:
//   (1) scores  S = (scale*Q) K^T                 (8x v_wmma_f32_16x16x32_f16)
//   (2) in-tile exclusive tail sums E = sp @ U_tri (4x wmma)
//   (3) output  O += P V                           (8x wmma)
// Each wave owns 32 t-rows so K/V register tiles feed two row-tiles.
// Elementwise path uses native v_exp_f32/v_log_f32 (VALU-bound kernel).
// ---------------------------------------------------------------------------

typedef __attribute__((ext_vector_type(16))) _Float16 v16h;
typedef __attribute__((ext_vector_type(2)))  _Float16 v2h;
typedef __attribute__((ext_vector_type(8)))  float    v8f;

constexpr int   Bsz = 2, Ssz = 2048, Hsz = 16, Dsz = 64;
constexpr int   SST   = 3 * Hsz * Dsz;   // 3072 floats between s steps in qkv
constexpr float SCALE = 1.5f / 8.0f;     // 1.5 / sqrt(64)
constexpr int   LROW  = 34;              // LDS f16 row stride (even, 17 dwords)

// Fast softplus on native transcendentals: max(x,0) + log(1 + exp(-|x|)).
// y = exp(-|x|) in (0,1]; log(1+y) via native log is exact enough here
// (absolute exponent error < 1e-8 once |x| > 18, dwarfed by f16 scores).
__device__ __forceinline__ float softplus_fast(float x) {
    return fmaxf(x, 0.0f) + __logf(1.0f + __expf(-fabsf(x)));
}

__global__ __launch_bounds__(128)
void sb_attn_fwd(const float* __restrict__ qkv, float* __restrict__ out) {
    // per-wave 32x32 f16 scratch tile (sp, then att), padded rows
    __shared__ _Float16 tl[4][32 * LROW];

    const int lane = threadIdx.x & 31;
    const int wid  = threadIdx.x >> 5;
    const int nl   = lane & 15;          // col-within-16 / row-within-16 id
    const int h2   = lane >> 4;          // wave half (0/1)

    const int b  = blockIdx.z;
    const int h  = blockIdx.y;
    const int t0 = (blockIdx.x * 4 + wid) * 32;    // this wave's 32 t-rows

    const float* baseQ = qkv + (size_t)b * Ssz * SST + (size_t)h * Dsz;
    const float* baseK = baseQ + Hsz * Dsz;        // qkv[...,1,h,:]
    const float* baseV = baseQ + 2 * Hsz * Dsz;    // qkv[...,2,h,:]

    // Constant strict-upper-triangular ones in B layout: U[n] maps 32 K-rows
    // to 16 cols (cols 16n..16n+15); entry(K, col) = (K > col). Exact in f16.
    v16h U[2];
#pragma unroll
    for (int n = 0; n < 2; ++n)
#pragma unroll
        for (int j = 0; j < 16; ++j)
            U[n][j] = (_Float16)(((j + 16 * h2) > (nl + 16 * n)) ? 1.0f : 0.0f);

    // ---- Q tiles (2 row-tiles x 64 d), pre-scaled, f16, WMMA A-layout ------
    // A 16x32 f16, contiguous k order used here (B operands match): lane<16 -> M=lane, K(j)=j ; lane>=16 -> K(j)=j+16
    v16h Aq[2][2];
#pragma unroll
    for (int u = 0; u < 2; ++u) {
        const float* qrow = baseQ + (size_t)(t0 + 16 * u + nl) * SST;
#pragma unroll
        for (int c = 0; c < 2; ++c)
#pragma unroll
            for (int j = 0; j < 16; ++j) {
                const int K = j + 16 * h2;
                Aq[u][c][j] = (_Float16)(SCALE * qrow[32 * c + K]);
            }
    }

    const v8f vzero = {};
    v8f O[2][4];                         // out accum (C/D layout), per row-tile
#pragma unroll
    for (int u = 0; u < 2; ++u)
#pragma unroll
        for (int g = 0; g < 4; ++g) O[u][g] = vzero;
    float R[2][8];                       // running softplus sum per owned row
#pragma unroll
    for (int u = 0; u < 2; ++u)
#pragma unroll
        for (int r = 0; r < 8; ++r) R[u][r] = 0.0f;

    // diagonal tile is sc == t0 (rows t0..t0+31 need s <= t0+30)
    for (int sc = t0; sc >= 0; sc -= 32) {
        // ---- K tile in B layout: 2 d-chunks x 2 s-subtiles ----------------
        v16h Bk[2][2];
#pragma unroll
        for (int n = 0; n < 2; ++n) {
            const float* krow =
                baseK + (size_t)(sc + 16 * n + nl) * SST + 16 * h2;
#pragma unroll
            for (int c = 0; c < 2; ++c)
#pragma unroll
                for (int j = 0; j < 16; ++j)
                    Bk[c][n][j] = (_Float16)krow[32 * c + j];
        }

        // ---- scores: S[u][n] = (scale*Q_u) . K  (K-dim 64 = 2 chained) ----
        v8f S[2][2];
#pragma unroll
        for (int u = 0; u < 2; ++u)
#pragma unroll
            for (int n = 0; n < 2; ++n) {
                v8f acc = vzero;
                acc = __builtin_amdgcn_wmma_f32_16x16x32_f16(false, Aq[u][0], false, Bk[0][n], (short)0, acc, false, false);
                acc = __builtin_amdgcn_wmma_f32_16x16x32_f16(false, Aq[u][1], false, Bk[1][n], (short)0, acc, false, false);
                S[u][n] = acc;
            }

        // ---- elementwise: softplus / logsigmoid; masked sp -> LDS (f16) ---
        // C/D layout: lane<16 -> col=lane, row=r ; lane>=16 -> col=lane-16, row=r+8
        float ls[2][2][8], sp0[2][8];
#pragma unroll
        for (int u = 0; u < 2; ++u)
#pragma unroll
            for (int r = 0; r < 8; ++r) {
                const int t   = t0 + 16 * u + 8 * h2 + r;
                const int row = 16 * u + 8 * h2 + r;
#pragma unroll
                for (int n = 0; n < 2; ++n) {
                    const float x   = S[u][n][r];
                    const float sp  = softplus_fast(x);
                    const bool  act = (sc + 16 * n + nl) < t;   // strict causal
                    const float spm = act ? sp : 0.0f;
                    ls[u][n][r] = x - sp;                       // logsigmoid(x)
                    tl[wid][row * LROW + 16 * n + nl] = (_Float16)spm;
                    if (n == 0) sp0[u][r] = spm;
                }
            }
        asm volatile("s_wait_dscnt 0" ::: "memory");   // same-wave DS RAW fence

        // ---- in-tile exclusive tail sums via WMMA: E = sp @ U -------------
        v8f E[2][2];
#pragma unroll
        for (int u = 0; u < 2; ++u) {
            v16h Asp;
#pragma unroll
            for (int j = 0; j < 16; j += 2) {
                const int K = j + 16 * h2;  // K,K+1 contiguous
                const v2h p = *(const v2h*)&tl[wid][(16 * u + nl) * LROW + K];
                Asp[j] = p.x; Asp[j + 1] = p.y;
            }
            E[u][0] = __builtin_amdgcn_wmma_f32_16x16x32_f16(false, Asp, false, U[0], (short)0, vzero, false, false);
            E[u][1] = __builtin_amdgcn_wmma_f32_16x16x32_f16(false, Asp, false, U[1], (short)0, vzero, false, false);
        }

        // ---- att = exp(logsig - tail - R); fold row totals into R ---------
#pragma unroll
        for (int u = 0; u < 2; ++u)
#pragma unroll
            for (int r = 0; r < 8; ++r) {
                const int t   = t0 + 16 * u + 8 * h2 + r;
                const int row = 16 * u + 8 * h2 + r;
                // full 32-col row total lives at col 0: E0 + sp0 there
                const float tot  = __shfl(E[u][0][r] + sp0[u][r], lane & 16, 32);
                const bool  a0   = (sc + nl) < t;
                const bool  a1   = (sc + 16 + nl) < t;
                // exponents are <= 0; native exp flushes large-negative to 0
                const float att0 = a0 ? __expf(ls[u][0][r] - E[u][0][r] - R[u][r]) : 0.0f;
                const float att1 = a1 ? __expf(ls[u][1][r] - E[u][1][r] - R[u][r]) : 0.0f;
                R[u][r] += tot;
                tl[wid][row * LROW + nl]      = (_Float16)att0;
                tl[wid][row * LROW + 16 + nl] = (_Float16)att1;
            }
        asm volatile("s_wait_dscnt 0" ::: "memory");   // same-wave DS RAW fence

        // ---- relayout P into WMMA A-layout (packed f16 pair loads) --------
        v16h Ap[2];
#pragma unroll
        for (int u = 0; u < 2; ++u)
#pragma unroll
            for (int j = 0; j < 16; j += 2) {
                const int K = j + 16 * h2;
                const v2h p = *(const v2h*)&tl[wid][(16 * u + nl) * LROW + K];
                Ap[u][j] = p.x; Ap[u][j + 1] = p.y;
            }

        // ---- O += P @ V : one V register tile feeds BOTH row-tiles --------
#pragma unroll
        for (int g = 0; g < 4; ++g) {
            v16h Bv;
            const float* vcol =
                baseV + (size_t)(sc + 16 * h2) * SST + 16 * g + nl;
#pragma unroll
            for (int j = 0; j < 16; ++j)
                Bv[j] = (_Float16)vcol[(size_t)j * SST];
            O[0][g] = __builtin_amdgcn_wmma_f32_16x16x32_f16(false, Ap[0], false, Bv, (short)0, O[0][g], false, false);
            O[1][g] = __builtin_amdgcn_wmma_f32_16x16x32_f16(false, Ap[1], false, Bv, (short)0, O[1][g], false, false);
        }
    }

    // ---- epilogue: out = O + exp(-R) * v_t --------------------------------
#pragma unroll
    for (int u = 0; u < 2; ++u)
#pragma unroll
        for (int r = 0; r < 8; ++r) {
            const int   t    = t0 + 16 * u + 8 * h2 + r;
            const float rem  = __expf(-R[u][r]);
            const float* vrow = baseV + (size_t)t * SST;
            float*       orow = out + (((size_t)b * Ssz + t) * Hsz + h) * Dsz;
#pragma unroll
            for (int g = 0; g < 4; ++g) {
                const int d = 16 * g + nl;
                orow[d] = O[u][g][r] + rem * vrow[d];
            }
        }
}

extern "C" void kernel_launch(void* const* d_in, const int* in_sizes, int n_in,
                              void* d_out, int out_size, void* d_ws, size_t ws_size,
                              hipStream_t stream) {
    (void)in_sizes; (void)n_in; (void)out_size; (void)d_ws; (void)ws_size;
    const float* qkv = (const float*)d_in[0];
    float*       out = (float*)d_out;
    dim3 grid(Ssz / 128, Hsz, Bsz);   // 4 waves/block, 32 t-rows per wave
    dim3 block(128);
    hipLaunchKernelGGL(sb_attn_fwd, grid, block, 0, stream, qkv, out);
}
